// TransformerBlock_15066745274389
// MI455X (gfx1250) — matrix operand order repaired and checked by reading, not compiled, not run on hardware
//
#include <hip/hip_runtime.h>
#include <hip/hip_bf16.h>
#include <math.h>

#define DIM 1024
#define SEQ 2048
#define BATCH 2
#define HEADS 16
#define HDIM 64
#define HIDDEN 4096
#define TOKENS (BATCH * SEQ) // 4096

typedef __bf16 bf16;
typedef __attribute__((ext_vector_type(16))) __bf16 v16bf;
typedef __attribute__((ext_vector_type(8)))  __bf16 v8bf;
typedef __attribute__((ext_vector_type(8)))  float  v8f;
typedef int v4i __attribute__((vector_size(16))); // matches builtin param type

// Detect gfx1250 async global->LDS copy support in this toolchain.
#if defined(__has_builtin)
#  if __has_builtin(__builtin_amdgcn_global_load_async_to_lds_b128)
#    define USE_ASYNC_LDS 1
#  endif
#endif
#ifndef USE_ASYNC_LDS
#  define USE_ASYNC_LDS 0
#endif

// ---------------------------------------------------------------------------
// WMMA fragment helpers (CDNA5 16-bit layouts, CDNA5 ISA 7.12.2)
// A 16x32: lane<16 -> row=lane, K {0..7, 16..23}; lane>=16 -> row=lane-16, K {8..15, 24..31}
// B 32x16 (from column-major storage: row n = output column, contiguous K):
//   lane<16 -> col=lane, K {0..7, 16..23}; lane>=16 -> col=lane-16, K {8..15, 24..31}
// C/D 16x16: vgpr v, lanes 0-15 -> (m=v, n=lane); lanes 16-31 -> (m=v+8, n=lane-16)
// ---------------------------------------------------------------------------

__device__ __forceinline__ v16bf frag_a(const bf16* base, int stride, int lane) {
  const int half = lane >> 4;
  const int r = lane & 15;
  const bf16* p = base + r * stride + half * 8;
  v8bf lo = *(const v8bf*)p;          // K = half*8 + 0..7
  v8bf hi = *(const v8bf*)(p + 16);   // K = 16 + half*8 + 0..7
  v16bf f;
#pragma unroll
  for (int i = 0; i < 8; ++i) { f[i] = lo[i]; f[i + 8] = hi[i]; }
  return f;
}

__device__ __forceinline__ v16bf frag_b(const bf16* base, int stride, int lane) {
  const int half = lane >> 4;
  const int c = lane & 15;
  const bf16* p = base + c * stride + half * 8;
  v8bf lo = *(const v8bf*)p;
  v8bf hi = *(const v8bf*)(p + 16);
  v16bf f;
#pragma unroll
  for (int i = 0; i < 8; ++i) { f[i] = lo[i]; f[i + 8] = hi[i]; }
  return f;
}

__device__ __forceinline__ v8f wmma_bf16(v16bf a, v16bf b, v8f c) {
  return __builtin_amdgcn_wmma_f32_16x16x32_bf16(false, a, false, b, (short)0, c,
                                                 false, false);
}

// 16-byte global -> LDS copy; async (ASYNCcnt) when the toolchain exposes it.
__device__ __forceinline__ void copy16_g2l(const bf16* g, bf16* l) {
#if USE_ASYNC_LDS
  __builtin_amdgcn_global_load_async_to_lds_b128(
      (__attribute__((address_space(1))) v4i*)g,
      (__attribute__((address_space(3))) v4i*)l, 0, 0);
#else
  *(v8bf*)l = *(const v8bf*)g;
#endif
}

__device__ __forceinline__ void wait_async_le8() {
#if USE_ASYNC_LDS
#  if __has_builtin(__builtin_amdgcn_s_wait_asynccnt)
  __builtin_amdgcn_s_wait_asynccnt(8);
#  else
  asm volatile("s_wait_asynccnt 0x8" ::: "memory");
#  endif
#endif
}

__device__ __forceinline__ void wait_async_0() {
#if USE_ASYNC_LDS
#  if __has_builtin(__builtin_amdgcn_s_wait_asynccnt)
  __builtin_amdgcn_s_wait_asynccnt(0);
#  else
  asm volatile("s_wait_asynccnt 0x0" ::: "memory");
#  endif
#endif
}

// ---------------------------------------------------------------------------
// Weight transpose + f32->bf16 convert:  W[K][N] -> Wt[N][K]
// ---------------------------------------------------------------------------
__global__ __launch_bounds__(256) void transpose_f32_to_bf16(
    const float* __restrict__ W, bf16* __restrict__ Wt, int K, int N) {
  __shared__ float tile[32][33];
  const int tx = threadIdx.x & 31, ty = threadIdx.x >> 5;
  const int k0 = blockIdx.x * 32, n0 = blockIdx.y * 32;
#pragma unroll
  for (int i = 0; i < 4; ++i)
    tile[ty + 8 * i][tx] = W[(size_t)(k0 + ty + 8 * i) * N + n0 + tx];
  __syncthreads();
#pragma unroll
  for (int i = 0; i < 4; ++i)
    Wt[(size_t)(n0 + ty + 8 * i) * K + k0 + tx] = (bf16)tile[tx][ty + 8 * i];
}

// ---------------------------------------------------------------------------
// LayerNorm (one 256-thread block per row of 1024) -> bf16 out
// ---------------------------------------------------------------------------
__global__ __launch_bounds__(256) void layernorm_bf16_kernel(
    const float* __restrict__ x, const float* __restrict__ g,
    const float* __restrict__ b, bf16* __restrict__ out) {
  const int row = blockIdx.x;
  const int tid = threadIdx.x;
  const float* xr = x + (size_t)row * DIM;
  float vals[4], sum = 0.f, sq = 0.f;
#pragma unroll
  for (int i = 0; i < 4; ++i) {
    float t = xr[tid + 256 * i];
    vals[i] = t; sum += t; sq += t * t;
  }
#pragma unroll
  for (int mask = 1; mask < 32; mask <<= 1) {
    sum += __shfl_xor(sum, mask, 32);
    sq  += __shfl_xor(sq,  mask, 32);
  }
  __shared__ float red[2][8];
  if ((tid & 31) == 0) { red[0][tid >> 5] = sum; red[1][tid >> 5] = sq; }
  __syncthreads();
  sum = 0.f; sq = 0.f;
#pragma unroll
  for (int w = 0; w < 8; ++w) { sum += red[0][w]; sq += red[1][w]; }
  const float mu  = sum * (1.0f / DIM);
  const float var = sq * (1.0f / DIM) - mu * mu;
  const float rstd = rsqrtf(var + 1e-5f);
#pragma unroll
  for (int i = 0; i < 4; ++i) {
    int c = tid + 256 * i;
    out[(size_t)row * DIM + c] = (bf16)((vals[i] - mu) * rstd * g[c] + b[c]);
  }
}

// ---------------------------------------------------------------------------
// GEMM:  C[M,N] = A[M,K](bf16) * B[K,N]  with B given as Bt[N,K] (bf16)
// Block tile 128x128, 8 waves in 2(M)x4(N), wave tile 64x32 (4x2 WMMA accs).
// K staged 64 at a time, double-buffered LDS, async global->LDS when available.
// EPI 0: +bias -> bf16 | EPI 1: +bias, GELU -> bf16 | EPI 2: +bias+resid -> f32
// ---------------------------------------------------------------------------
template <int EPI>
__global__ __launch_bounds__(256) void gemm_bf16_kernel(
    const bf16* __restrict__ A, const bf16* __restrict__ Bt,
    const float* __restrict__ bias, const float* __restrict__ resid,
    float* __restrict__ outF, bf16* __restrict__ outB, int M, int N, int K) {
  __shared__ bf16 sA[2][128 * 64]; // 2 x 16 KB
  __shared__ bf16 sB[2][128 * 64]; // 2 x 16 KB
  const int tid  = threadIdx.x;
  const int wave = tid >> 5, lane = tid & 31;
  const int waveM = wave & 1, waveN = wave >> 1;   // 2 x 4 wave grid
  const int half = lane >> 4, nn = lane & 15;
  const int tileM = blockIdx.x * 128, tileN = blockIdx.y * 128;
  v8f acc[4][2] = {};

  // staging split: each thread moves 4x16B of A and 4x16B of B per 64-K panel
  const int r = tid >> 1, c = (tid & 1) * 32;
  const bf16* gA = A  + (size_t)(tileM + r) * K + c;
  const bf16* gB = Bt + (size_t)(tileN + r) * K + c;

  // prologue: stage panel 0 into buffer 0
#pragma unroll
  for (int i = 0; i < 4; ++i) copy16_g2l(gA + i * 8, &sA[0][r * 64 + c + i * 8]);
#pragma unroll
  for (int i = 0; i < 4; ++i) copy16_g2l(gB + i * 8, &sB[0][r * 64 + c + i * 8]);

  int cur = 0;
  for (int k0 = 0; k0 < K; k0 += 64) {
    if (k0 + 64 < K) { // stage next panel into the other buffer
      const int nxt = cur ^ 1;
#pragma unroll
      for (int i = 0; i < 4; ++i)
        copy16_g2l(gA + k0 + 64 + i * 8, &sA[nxt][r * 64 + c + i * 8]);
#pragma unroll
      for (int i = 0; i < 4; ++i)
        copy16_g2l(gB + k0 + 64 + i * 8, &sB[nxt][r * 64 + c + i * 8]);
      wait_async_le8(); // the 8 older ops (current panel) have landed
    } else {
      wait_async_0();
    }
    __syncthreads();

#pragma unroll
    for (int kk = 0; kk < 64; kk += 32) {
      v16bf a[4], b[2];
#pragma unroll
      for (int i = 0; i < 4; ++i)
        a[i] = frag_a(&sA[cur][(waveM * 64 + i * 16) * 64 + kk], 64, lane);
#pragma unroll
      for (int j = 0; j < 2; ++j)
        b[j] = frag_b(&sB[cur][(waveN * 32 + j * 16) * 64 + kk], 64, lane);
#pragma unroll
      for (int i = 0; i < 4; ++i)
#pragma unroll
        for (int j = 0; j < 2; ++j)
          acc[i][j] = wmma_bf16(a[i], b[j], acc[i][j]);
    }
    __syncthreads(); // everyone done reading `cur` before it is overwritten
    cur ^= 1;
  }

#pragma unroll
  for (int i = 0; i < 4; ++i)
#pragma unroll
    for (int j = 0; j < 2; ++j)
#pragma unroll
      for (int vv = 0; vv < 8; ++vv) {
        const int m = tileM + waveM * 64 + i * 16 + vv + 8 * half;
        const int n = tileN + waveN * 32 + j * 16 + nn;
        float val = acc[i][j][vv] + bias[n];
        if constexpr (EPI == 1)
          val = 0.5f * val * (1.0f + erff(val * 0.70710678118654752f));
        if constexpr (EPI == 2)
          outF[(size_t)m * N + n] = val + resid[(size_t)m * N + n];
        else
          outB[(size_t)m * N + n] = (bf16)val;
      }
}

// ---------------------------------------------------------------------------
// Flash attention: one block per (128-query tile, head, batch). 8 waves, each
// wave owns 16 query rows. q/k/v/o are [B*SEQ, DIM] bf16, head = 64-col slice.
// ---------------------------------------------------------------------------
__global__ __launch_bounds__(256) void attention_kernel(
    const bf16* __restrict__ q, const bf16* __restrict__ k,
    const bf16* __restrict__ v, bf16* __restrict__ o) {
  __shared__ bf16 sQ[128 * 64];   // 16 KB  queries, row-major [q][d]
  __shared__ bf16 sK[64 * 64];    //  8 KB  keys,    row-major [key][d]
  __shared__ bf16 sVt[64 * 64];   //  8 KB  values transposed [d][key]
  __shared__ bf16 sP[8][16 * 64]; // 16 KB  per-wave probs [qrow][key]

  const int tid  = threadIdx.x;
  const int wave = tid >> 5, lane = tid & 31;
  const int half = lane >> 4, nn = lane & 15;
  const int qb = blockIdx.x * 128;
  const size_t baseOff = (size_t)blockIdx.z * SEQ * DIM + blockIdx.y * HDIM;
  const bf16* qB = q + baseOff;
  const bf16* kB = k + baseOff;
  const bf16* vB = v + baseOff;
  bf16* oB = o + baseOff;

  { // stage Q tile once
    const int r = tid >> 1, c = (tid & 1) * 32;
    const bf16* g = qB + (size_t)(qb + r) * DIM + c;
#pragma unroll
    for (int i = 0; i < 4; ++i)
      *(v8bf*)&sQ[r * 64 + c + i * 8] = *(const v8bf*)(g + i * 8);
  }

  float m_row[8], l_row[8];
#pragma unroll
  for (int i = 0; i < 8; ++i) { m_row[i] = -1e30f; l_row[i] = 0.f; }
  v8f accO[4] = {};
  const float scale = 0.125f; // 1/sqrt(64)

  for (int kb0 = 0; kb0 < SEQ; kb0 += 64) {
    { // stage K block, row-major
      const int r = tid >> 2, c = (tid & 3) * 16;
      const bf16* g = kB + (size_t)(kb0 + r) * DIM + c;
      *(v8bf*)&sK[r * 64 + c]     = *(const v8bf*)g;
      *(v8bf*)&sK[r * 64 + c + 8] = *(const v8bf*)(g + 8);
    }
    { // stage V block transposed: sVt[d][key]
      const int key = tid >> 2, d0 = (tid & 3) * 16;
      const bf16* g = vB + (size_t)(kb0 + key) * DIM + d0;
#pragma unroll
      for (int e = 0; e < 16; ++e) sVt[(d0 + e) * 64 + key] = g[e];
    }
    __syncthreads();

    // S = Q_wave(16x64) * K^T(64x64keys): 4 col-tiles x 2 K-steps
    v8f s[4] = {};
#pragma unroll
    for (int kk = 0; kk < 64; kk += 32) {
      v16bf a = frag_a(&sQ[(wave * 16) * 64 + kk], 64, lane);
#pragma unroll
      for (int j = 0; j < 4; ++j) {
        v16bf bfr = frag_b(&sK[(j * 16) * 64 + kk], 64, lane);
        s[j] = wmma_bf16(a, bfr, s[j]);
      }
    }

    // online softmax per query row (row = vv + 8*half, spread over 16 lanes)
#pragma unroll
    for (int vv = 0; vv < 8; ++vv) {
      float t = fmaxf(fmaxf(s[0][vv], s[1][vv]), fmaxf(s[2][vv], s[3][vv]));
#pragma unroll
      for (int mask = 1; mask < 16; mask <<= 1)
        t = fmaxf(t, __shfl_xor(t, mask, 32));
      const float mn = fmaxf(m_row[vv], t * scale);
      const float corr = expf(m_row[vv] - mn);
      float rs = 0.f;
#pragma unroll
      for (int j = 0; j < 4; ++j) {
        float p = expf(s[j][vv] * scale - mn);
        rs += p;
        sP[wave][(vv + 8 * half) * 64 + j * 16 + nn] = (bf16)p;
      }
#pragma unroll
      for (int mask = 1; mask < 16; mask <<= 1)
        rs += __shfl_xor(rs, mask, 32);
      l_row[vv] = l_row[vv] * corr + rs;
      m_row[vv] = mn;
#pragma unroll
      for (int j = 0; j < 4; ++j) accO[j][vv] = accO[j][vv] * corr;
    }
    __syncthreads();

    // O += P(16x64keys) * V(64keys x 64d)
#pragma unroll
    for (int kk = 0; kk < 64; kk += 32) {
      v16bf a = frag_a(&sP[wave][kk], 64, lane);
#pragma unroll
      for (int j = 0; j < 4; ++j) {
        v16bf bfr = frag_b(&sVt[(j * 16) * 64 + kk], 64, lane);
        accO[j] = wmma_bf16(a, bfr, accO[j]);
      }
    }
    __syncthreads();
  }

#pragma unroll
  for (int vv = 0; vv < 8; ++vv) {
    const float inv = 1.0f / l_row[vv];
    const int row = qb + wave * 16 + vv + 8 * half;
#pragma unroll
    for (int j = 0; j < 4; ++j)
      oB[(size_t)row * DIM + j * 16 + nn] = (bf16)(accO[j][vv] * inv);
  }
}

// ---------------------------------------------------------------------------
// Host-side orchestration
// ---------------------------------------------------------------------------
extern "C" void kernel_launch(void* const* d_in, const int* in_sizes, int n_in,
                              void* d_out, int out_size, void* d_ws, size_t ws_size,
                              hipStream_t stream) {
  const float* x   = (const float*)d_in[0];
  const float* g1  = (const float*)d_in[1];
  const float* b1  = (const float*)d_in[2];
  const float* Wq  = (const float*)d_in[3];
  const float* bq  = (const float*)d_in[4];
  const float* Wk  = (const float*)d_in[5];
  const float* bk  = (const float*)d_in[6];
  const float* Wv  = (const float*)d_in[7];
  const float* bv  = (const float*)d_in[8];
  const float* Wo  = (const float*)d_in[9];
  const float* bo  = (const float*)d_in[10];
  const float* g2  = (const float*)d_in[11];
  const float* b2  = (const float*)d_in[12];
  const float* W1  = (const float*)d_in[13];
  const float* bf1 = (const float*)d_in[14];
  const float* W2  = (const float*)d_in[15];
  const float* bf2 = (const float*)d_in[16];
  float* out = (float*)d_out;

  char* ws = (char*)d_ws;
  size_t off = 0;
  auto alloc = [&](size_t bytes) -> void* {
    void* p = ws + off;
    off += (bytes + 255) & ~(size_t)255;
    return p;
  };
  bf16* WqT = (bf16*)alloc((size_t)DIM * DIM * 2);
  bf16* WkT = (bf16*)alloc((size_t)DIM * DIM * 2);
  bf16* WvT = (bf16*)alloc((size_t)DIM * DIM * 2);
  bf16* WoT = (bf16*)alloc((size_t)DIM * DIM * 2);
  bf16* W1T = (bf16*)alloc((size_t)DIM * HIDDEN * 2);
  bf16* W2T = (bf16*)alloc((size_t)HIDDEN * DIM * 2);
  bf16* h1  = (bf16*)alloc((size_t)TOKENS * DIM * 2);
  bf16* qb  = (bf16*)alloc((size_t)TOKENS * DIM * 2);
  bf16* kb  = (bf16*)alloc((size_t)TOKENS * DIM * 2);
  bf16* vb  = (bf16*)alloc((size_t)TOKENS * DIM * 2);
  bf16* ob  = (bf16*)alloc((size_t)TOKENS * DIM * 2);
  float* x1 = (float*)alloc((size_t)TOKENS * DIM * 4);
  bf16* h2  = (bf16*)alloc((size_t)TOKENS * DIM * 2);
  bf16* ffh = (bf16*)alloc((size_t)TOKENS * HIDDEN * 2);
  (void)ws_size; (void)in_sizes; (void)n_in; (void)out_size;

  const dim3 tb(256);

  // weights -> bf16 transposed
  transpose_f32_to_bf16<<<dim3(DIM / 32, DIM / 32), tb, 0, stream>>>(Wq, WqT, DIM, DIM);
  transpose_f32_to_bf16<<<dim3(DIM / 32, DIM / 32), tb, 0, stream>>>(Wk, WkT, DIM, DIM);
  transpose_f32_to_bf16<<<dim3(DIM / 32, DIM / 32), tb, 0, stream>>>(Wv, WvT, DIM, DIM);
  transpose_f32_to_bf16<<<dim3(DIM / 32, DIM / 32), tb, 0, stream>>>(Wo, WoT, DIM, DIM);
  transpose_f32_to_bf16<<<dim3(DIM / 32, HIDDEN / 32), tb, 0, stream>>>(W1, W1T, DIM, HIDDEN);
  transpose_f32_to_bf16<<<dim3(HIDDEN / 32, DIM / 32), tb, 0, stream>>>(W2, W2T, HIDDEN, DIM);

  // LN1
  layernorm_bf16_kernel<<<TOKENS, tb, 0, stream>>>(x, g1, b1, h1);

  // QKV projections
  gemm_bf16_kernel<0><<<dim3(TOKENS / 128, DIM / 128), tb, 0, stream>>>(
      h1, WqT, bq, nullptr, nullptr, qb, TOKENS, DIM, DIM);
  gemm_bf16_kernel<0><<<dim3(TOKENS / 128, DIM / 128), tb, 0, stream>>>(
      h1, WkT, bk, nullptr, nullptr, kb, TOKENS, DIM, DIM);
  gemm_bf16_kernel<0><<<dim3(TOKENS / 128, DIM / 128), tb, 0, stream>>>(
      h1, WvT, bv, nullptr, nullptr, vb, TOKENS, DIM, DIM);

  // attention
  attention_kernel<<<dim3(SEQ / 128, HEADS, BATCH), tb, 0, stream>>>(qb, kb, vb, ob);

  // out projection + residual -> x1 (f32)
  gemm_bf16_kernel<2><<<dim3(TOKENS / 128, DIM / 128), tb, 0, stream>>>(
      ob, WoT, bo, x, x1, nullptr, TOKENS, DIM, DIM);

  // LN2
  layernorm_bf16_kernel<<<TOKENS, tb, 0, stream>>>(x1, g2, b2, h2);

  // FFN1 + GELU
  gemm_bf16_kernel<1><<<dim3(TOKENS / 128, HIDDEN / 128), tb, 0, stream>>>(
      h2, W1T, bf1, nullptr, nullptr, ffh, TOKENS, HIDDEN, DIM);

  // FFN2 + residual -> out (f32)
  gemm_bf16_kernel<2><<<dim3(TOKENS / 128, DIM / 128), tb, 0, stream>>>(
      ffh, W2T, bf2, x1, out, nullptr, TOKENS, DIM, HIDDEN);
}
